// Model_14465449853445
// MI455X (gfx1250) — matrix operand order repaired and checked by reading, not compiled, not run on hardware
//
#include <hip/hip_runtime.h>
#include <hip/hip_bf16.h>

typedef __attribute__((ext_vector_type(16))) __bf16 v16bf;
typedef __attribute__((ext_vector_type(8)))  float  v8f;

// float -> bf16 via native conversion (RNE)
__device__ __forceinline__ __bf16 f2bf(float f) { return (__bf16)f; }

// two floats -> packed 2xbf16 in one dword (lo = first)
__device__ __forceinline__ unsigned pack2bf(float lo, float hi) {
    unsigned short a = __builtin_bit_cast(unsigned short, (__bf16)lo);
    unsigned short b = __builtin_bit_cast(unsigned short, (__bf16)hi);
    return (unsigned)a | ((unsigned)b << 16);
}

// 32-bit LDS offset (addrspace(3) value) of a generic pointer to __shared__ memory
__device__ __forceinline__ unsigned lds_offset(const void* p) {
    return (unsigned)(unsigned long long)(const __attribute__((address_space(3))) void*)p;
}

// CDNA5 async global->LDS copy of 16 bytes (tracked by ASYNCcnt)
__device__ __forceinline__ void async_copy16(const void* gsrc, unsigned ldsoff) {
    asm volatile("global_load_async_to_lds_b128 %0, %1, off"
                 :: "v"(ldsoff), "v"(gsrc) : "memory");
}
__device__ __forceinline__ void wait_async0() {
    asm volatile("s_wait_asynccnt 0x0" ::: "memory");
}

// A fragment: 16x32 bf16, row-major LDS tile (lda elements per row), k-step offset ks.
__device__ __forceinline__ v16bf load_a_frag(const __bf16* A, int lda, int lane, int ks) {
    const int m  = lane & 15;
    const int kh = (lane >> 4) << 3;      // 0 or 8
    v16bf a;
#pragma unroll
    for (int i = 0; i < 16; ++i) {
        int k = ks + kh + ((i >> 3) << 4) + (i & 7);
        a[i] = A[m * lda + k];
    }
    return a;
}

// B fragment: 32x16 bf16 from a TRANSPOSED LDS tile Bt[n][k].
__device__ __forceinline__ v16bf load_b_frag(const __bf16* Bt, int ldb, int lane, int n0, int ks) {
    const int n  = n0 + (lane & 15);
    const int kb = ks + ((lane >> 4) << 3); // 0 or 8
    v16bf b;
#pragma unroll
    for (int i = 0; i < 16; ++i) b[i] = Bt[n * ldb + kb + ((i >> 3) << 4) + (i & 7)];
    return b;
}

// ---------------------------------------------------------------------------
// Input fusion, recipe side: out[N,64] = x[N,K] @ W[K,64] + bias + emb[nid]
// K must be a multiple of 32. Block = 128 threads = 4 waves, 16 rows.
// A tile staged fp32 via async global->LDS (double buffered); W tile staged
// as packed 2xbf16 [n][k/2] via float4 loads, stores deferred past the WMMA.
// ---------------------------------------------------------------------------
__global__ void fuse_recipe_kernel(const float* __restrict__ x, const float* __restrict__ W,
                                   const float* __restrict__ bias, const float* __restrict__ emb,
                                   const int* __restrict__ nid, float* __restrict__ out,
                                   int N, int K) {
    __shared__ float    sAf[2][16][32];   // fp32 A tiles (async-filled)
    __shared__ unsigned sBu[2][64][16];   // W tiles, transposed, packed 2xbf16 along k
    const int tid = threadIdx.x, lane = tid & 31, wave = tid >> 5;
    const int row0 = blockIdx.x * 16;

    // per-thread A async-copy coordinates: one 16B chunk per tile
    const int ar = tid >> 3;            // row 0..15
    const int ac = (tid & 7) << 2;      // col 0,4,..,28
    int arow = row0 + ar; if (arow >= N) arow = N - 1;
    const float* agsrc = x + (size_t)arow * K + ac;

    // B staging coords (per j): i = tid + 128*j ; kk2 = i>>4 (0..15), n4 = (i&15)*4
    // ---- prologue: tile 0 ----
    async_copy16(agsrc, lds_offset(&sAf[0][ar][ac]));
    {
        float4 w0[2], w1[2];
#pragma unroll
        for (int j = 0; j < 2; ++j) {
            int i = tid + 128 * j, kk2 = i >> 4, n4 = (i & 15) << 2;
            w0[j] = *(const float4*)(W + (size_t)(2 * kk2)     * 64 + n4);
            w1[j] = *(const float4*)(W + (size_t)(2 * kk2 + 1) * 64 + n4);
        }
#pragma unroll
        for (int j = 0; j < 2; ++j) {
            int i = tid + 128 * j, kk2 = i >> 4, n4 = (i & 15) << 2;
            sBu[0][n4 + 0][kk2] = pack2bf(w0[j].x, w1[j].x);
            sBu[0][n4 + 1][kk2] = pack2bf(w0[j].y, w1[j].y);
            sBu[0][n4 + 2][kk2] = pack2bf(w0[j].z, w1[j].z);
            sBu[0][n4 + 3][kk2] = pack2bf(w0[j].w, w1[j].w);
        }
    }
    wait_async0();
    __syncthreads();

    v8f c = {};
    for (int k0 = 0; k0 < K; k0 += 32) {
        const int cur = (k0 >> 5) & 1, nxt = cur ^ 1;
        const bool more = (k0 + 32) < K;      // block-uniform
        float4 w0[2], w1[2];
        if (more) {
            async_copy16(agsrc + k0 + 32, lds_offset(&sAf[nxt][ar][ac]));
#pragma unroll
            for (int j = 0; j < 2; ++j) {
                int i = tid + 128 * j, kk2 = i >> 4, n4 = (i & 15) << 2;
                w0[j] = *(const float4*)(W + (size_t)(k0 + 32 + 2 * kk2)     * 64 + n4);
                w1[j] = *(const float4*)(W + (size_t)(k0 + 32 + 2 * kk2 + 1) * 64 + n4);
            }
        }
        // compute on current tile: convert fp32 A to bf16 during fragment build
        v16bf a;
        {
            const int m  = lane & 15;
            const int kh = (lane >> 4) << 3;
#pragma unroll
            for (int i = 0; i < 16; ++i) {
                int k = kh + ((i >> 3) << 4) + (i & 7);
                a[i] = f2bf(sAf[cur][m][k]);
            }
        }
        v16bf b = load_b_frag((const __bf16*)&sBu[cur][0][0], 32, lane, wave * 16, 0);
        c = __builtin_amdgcn_wmma_f32_16x16x32_bf16(false, a, false, b, (short)0, c, false, false);
        if (more) {
#pragma unroll
            for (int j = 0; j < 2; ++j) {
                int i = tid + 128 * j, kk2 = i >> 4, n4 = (i & 15) << 2;
                sBu[nxt][n4 + 0][kk2] = pack2bf(w0[j].x, w1[j].x);
                sBu[nxt][n4 + 1][kk2] = pack2bf(w0[j].y, w1[j].y);
                sBu[nxt][n4 + 2][kk2] = pack2bf(w0[j].z, w1[j].z);
                sBu[nxt][n4 + 3][kk2] = pack2bf(w0[j].w, w1[j].w);
            }
            wait_async0();
            __syncthreads();
        }
    }

    const int n = wave * 16 + (lane & 15);
    const int mOff = (lane >> 4) * 8;
#pragma unroll
    for (int v = 0; v < 8; ++v) {
        int r = row0 + mOff + v;
        if (r < N) out[(size_t)r * 64 + n] = c[v] + bias[n] + emb[(size_t)nid[r] * 64 + n];
    }
}

// ---------------------------------------------------------------------------
// Input fusion, user side: K (=58) padded to 64 with zeros; W staged whole.
// ---------------------------------------------------------------------------
__global__ void fuse_user_kernel(const float* __restrict__ x, const float* __restrict__ W,
                                 const float* __restrict__ bias, const float* __restrict__ emb,
                                 const int* __restrict__ nid, float* __restrict__ out,
                                 int N, int K) {
    __shared__ __bf16 sA[16][64];
    __shared__ __bf16 sB[64][64];   // transposed [n][k], zero-padded k >= K
    const int tid = threadIdx.x, lane = tid & 31, wave = tid >> 5;
    const int row0 = blockIdx.x * 16;
    const __bf16 bz = __builtin_bit_cast(__bf16, (unsigned short)0);
    for (int i = tid; i < 64 * 64; i += 128) {
        int kk = i >> 6, n = i & 63;
        sB[n][kk] = (kk < K) ? f2bf(W[(size_t)kk * 64 + n]) : bz;
    }
    for (int i = tid; i < 16 * 64; i += 128) {
        int r = i >> 6, kk = i & 63;
        int row = row0 + r; if (row >= N) row = N - 1;
        sA[r][kk] = (kk < K) ? f2bf(x[(size_t)row * K + kk]) : bz;
    }
    __syncthreads();
    v8f c = {};
#pragma unroll
    for (int ks = 0; ks < 64; ks += 32) {
        v16bf a = load_a_frag(&sA[0][0], 64, lane, ks);
        v16bf b = load_b_frag(&sB[0][0], 64, lane, wave * 16, ks);
        c = __builtin_amdgcn_wmma_f32_16x16x32_bf16(false, a, false, b, (short)0, c, false, false);
    }
    const int n = wave * 16 + (lane & 15);
    const int mOff = (lane >> 4) * 8;
#pragma unroll
    for (int v = 0; v < 8; ++v) {
        int r = row0 + mOff + v;
        if (r < N) out[(size_t)r * 64 + n] = c[v] + bias[n] + emb[(size_t)nid[r] * 64 + n];
    }
}

// ---------------------------------------------------------------------------
// Edge degree counts (layer-invariant, computed once per call).
// ---------------------------------------------------------------------------
__global__ void degree_kernel(const int* __restrict__ esrc, const int* __restrict__ edst,
                              float* __restrict__ degU, float* __restrict__ degR, int E) {
    int e = blockIdx.x * blockDim.x + threadIdx.x;
    if (e >= E) return;
    unsafeAtomicAdd(&degU[esrc[e]], 1.0f);
    unsafeAtomicAdd(&degR[edst[e]], 1.0f);
}

// ---------------------------------------------------------------------------
// Segment-sum scatter: 16 threads per edge, float4 gather + 4 HW f32 atomics.
// One lane per edge prefetches the index streams ahead (global_prefetch_b8).
// ---------------------------------------------------------------------------
__global__ void scatter_sum_kernel(const float* __restrict__ hsrc,
                                   const int* __restrict__ esrc, const int* __restrict__ edst,
                                   float* __restrict__ sum, int E) {
    int gid = blockIdx.x * blockDim.x + threadIdx.x;
    int e = gid >> 4;
    if (e >= E) return;
    int j = (gid & 15) << 2;
    if ((gid & 15) == 0 && (e + 2048) < E) {
        __builtin_prefetch(esrc + e + 2048, 0, 1);
        __builtin_prefetch(edst + e + 2048, 0, 1);
    }
    int s = esrc[e], d = edst[e];
    const float4 v = *(const float4*)(hsrc + (size_t)s * 64 + j);
    float* p = sum + (size_t)d * 64 + j;
    unsafeAtomicAdd(p + 0, v.x);
    unsafeAtomicAdd(p + 1, v.y);
    unsafeAtomicAdd(p + 2, v.z);
    unsafeAtomicAdd(p + 3, v.w);
}

// ---------------------------------------------------------------------------
// SAGE combine: out = (sum/max(deg,1)) @ Wl + bl + xdst @ Wr, optional ReLU.
// Two chained WMMA accumulations (K=64 -> 2 k-steps each) into one fp32 acc.
// Weight/row tiles staged as packed 2xbf16 along k.
// ---------------------------------------------------------------------------
__global__ void sage_combine_kernel(const float* __restrict__ sum, const float* __restrict__ deg,
                                    const float* __restrict__ xdst,
                                    const float* __restrict__ Wl, const float* __restrict__ bl,
                                    const float* __restrict__ Wr,
                                    float* __restrict__ out, int N, int relu) {
    __shared__ unsigned sWlu[64][32];  // [n][k/2] packed 2xbf16
    __shared__ unsigned sWru[64][32];  // [n][k/2]
    __shared__ unsigned sMu[16][32];   // mean rows, packed along k
    __shared__ unsigned sXu[16][32];   // x_dst rows, packed along k
    const int tid = threadIdx.x, lane = tid & 31, wave = tid >> 5;
    const int row0 = blockIdx.x * 16;
    // weights: (kk2, n4) pairs -> 2 float4 loads + 4 packed b32 stores each;
    // 32 kk2 * 16 n4 = 512 items per matrix, 4 per thread
#pragma unroll
    for (int j = 0; j < 4; ++j) {
        int i = tid + 128 * j, kk2 = i >> 4, n4 = (i & 15) << 2;
        float4 a0 = *(const float4*)(Wl + (size_t)(2 * kk2)     * 64 + n4);
        float4 a1 = *(const float4*)(Wl + (size_t)(2 * kk2 + 1) * 64 + n4);
        float4 b0 = *(const float4*)(Wr + (size_t)(2 * kk2)     * 64 + n4);
        float4 b1 = *(const float4*)(Wr + (size_t)(2 * kk2 + 1) * 64 + n4);
        sWlu[n4 + 0][kk2] = pack2bf(a0.x, a1.x);
        sWlu[n4 + 1][kk2] = pack2bf(a0.y, a1.y);
        sWlu[n4 + 2][kk2] = pack2bf(a0.z, a1.z);
        sWlu[n4 + 3][kk2] = pack2bf(a0.w, a1.w);
        sWru[n4 + 0][kk2] = pack2bf(b0.x, b1.x);
        sWru[n4 + 1][kk2] = pack2bf(b0.y, b1.y);
        sWru[n4 + 2][kk2] = pack2bf(b0.z, b1.z);
        sWru[n4 + 3][kk2] = pack2bf(b0.w, b1.w);
    }
    // mean + x_dst rows: 256 float4s each, 2 per thread; k contiguous -> packed pairs
#pragma unroll
    for (int j = 0; j < 2; ++j) {
        int i = tid + 128 * j, r = i >> 4, c4 = (i & 15) << 2;
        int row = row0 + r; if (row >= N) row = N - 1;
        float dcnt = deg[row]; dcnt = dcnt > 1.0f ? dcnt : 1.0f;
        float inv = 1.0f / dcnt;
        float4 sv = *(const float4*)(sum  + (size_t)row * 64 + c4);
        float4 xv = *(const float4*)(xdst + (size_t)row * 64 + c4);
        sMu[r][(c4 >> 1) + 0] = pack2bf(sv.x * inv, sv.y * inv);
        sMu[r][(c4 >> 1) + 1] = pack2bf(sv.z * inv, sv.w * inv);
        sXu[r][(c4 >> 1) + 0] = pack2bf(xv.x, xv.y);
        sXu[r][(c4 >> 1) + 1] = pack2bf(xv.z, xv.w);
    }
    __syncthreads();
    v8f c = {};
#pragma unroll
    for (int ks = 0; ks < 64; ks += 32) {
        v16bf a = load_a_frag((const __bf16*)&sMu[0][0], 64, lane, ks);
        v16bf b = load_b_frag((const __bf16*)&sWlu[0][0], 64, lane, wave * 16, ks);
        c = __builtin_amdgcn_wmma_f32_16x16x32_bf16(false, a, false, b, (short)0, c, false, false);
    }
#pragma unroll
    for (int ks = 0; ks < 64; ks += 32) {
        v16bf a = load_a_frag((const __bf16*)&sXu[0][0], 64, lane, ks);
        v16bf b = load_b_frag((const __bf16*)&sWru[0][0], 64, lane, wave * 16, ks);
        c = __builtin_amdgcn_wmma_f32_16x16x32_bf16(false, a, false, b, (short)0, c, false, false);
    }
    const int n = wave * 16 + (lane & 15);
    const int mOff = (lane >> 4) * 8;
#pragma unroll
    for (int v = 0; v < 8; ++v) {
        int r = row0 + mOff + v;
        if (r < N) {
            float val = c[v] + bl[n];
            if (relu) val = val > 0.0f ? val : 0.0f;
            out[(size_t)r * 64 + n] = val;
        }
    }
}

extern "C" void kernel_launch(void* const* d_in, const int* in_sizes, int n_in,
                              void* d_out, int out_size, void* d_ws, size_t ws_size,
                              hipStream_t stream) {
    const float* x_user   = (const float*)d_in[0];
    const float* x_recipe = (const float*)d_in[1];
    const int*   nid_u    = (const int*)d_in[2];
    const int*   nid_r    = (const int*)d_in[3];
    const int*   e_src    = (const int*)d_in[4];
    const int*   e_dst    = (const int*)d_in[5];
    const float* W_ul = (const float*)d_in[6];
    const float* b_ul = (const float*)d_in[7];
    const float* W_rl = (const float*)d_in[8];
    const float* b_rl = (const float*)d_in[9];
    const float* emb_u = (const float*)d_in[10];
    const float* emb_r = (const float*)d_in[11];
    const float* Wl1_ur = (const float*)d_in[12];
    const float* bl1_ur = (const float*)d_in[13];
    const float* Wr1_ur = (const float*)d_in[14];
    const float* Wl1_ru = (const float*)d_in[15];
    const float* bl1_ru = (const float*)d_in[16];
    const float* Wr1_ru = (const float*)d_in[17];
    const float* Wl2_ur = (const float*)d_in[18];
    const float* bl2_ur = (const float*)d_in[19];
    const float* Wr2_ur = (const float*)d_in[20];
    const float* Wl2_ru = (const float*)d_in[21];
    const float* bl2_ru = (const float*)d_in[22];
    const float* Wr2_ru = (const float*)d_in[23];

    const int U  = in_sizes[2];
    const int R  = in_sizes[3];
    const int E  = in_sizes[4];
    const int KU = in_sizes[0] / U;   // 58
    const int KR = in_sizes[1] / R;   // 768

    float* ws = (float*)d_ws;
    size_t off = 0;
    float* hu    = ws + off; off += (size_t)U * 64;
    float* hr    = ws + off; off += (size_t)R * 64;
    float* hu1   = ws + off; off += (size_t)U * 64;
    float* hr1   = ws + off; off += (size_t)R * 64;
    float* sum_u = ws + off; off += (size_t)U * 64;   // sum_u and sum_r contiguous
    float* sum_r = ws + off; off += (size_t)R * 64;
    float* deg_u = ws + off; off += (size_t)U;        // deg_u and deg_r contiguous
    float* deg_r = ws + off; off += (size_t)R;
    if (ws_size < off * sizeof(float)) return;

    float* out_u = (float*)d_out;
    float* out_r = out_u + (size_t)U * 64;

    const dim3 blk(128);
    const int gu = (U + 15) / 16;
    const int gr = (R + 15) / 16;
    const int gE1 = (E + 255) / 256;
    const int gE16 = (int)(((size_t)E * 16 + 255) / 256);

    // degrees (layer-invariant)
    hipMemsetAsync(deg_u, 0, (size_t)(U + R) * sizeof(float), stream);
    degree_kernel<<<gE1, 256, 0, stream>>>(e_src, e_dst, deg_u, deg_r, E);

    // input fusion
    fuse_user_kernel<<<gu, blk, 0, stream>>>(x_user, W_ul, b_ul, emb_u, nid_u, hu, U, KU);
    fuse_recipe_kernel<<<gr, blk, 0, stream>>>(x_recipe, W_rl, b_rl, emb_r, nid_r, hr, R, KR);

    // layer 1: scatter + combine (+ReLU)
    hipMemsetAsync(sum_u, 0, (size_t)(U + R) * 64 * sizeof(float), stream);
    scatter_sum_kernel<<<gE16, 256, 0, stream>>>(hr, e_dst, e_src, sum_u, E); // recipes -> users
    scatter_sum_kernel<<<gE16, 256, 0, stream>>>(hu, e_src, e_dst, sum_r, E); // users -> recipes
    sage_combine_kernel<<<gu, blk, 0, stream>>>(sum_u, deg_u, hu, Wl1_ru, bl1_ru, Wr1_ru, hu1, U, 1);
    sage_combine_kernel<<<gr, blk, 0, stream>>>(sum_r, deg_r, hr, Wl1_ur, bl1_ur, Wr1_ur, hr1, R, 1);

    // layer 2: scatter + combine (no ReLU), write straight to d_out
    hipMemsetAsync(sum_u, 0, (size_t)(U + R) * 64 * sizeof(float), stream);
    scatter_sum_kernel<<<gE16, 256, 0, stream>>>(hr1, e_dst, e_src, sum_u, E);
    scatter_sum_kernel<<<gE16, 256, 0, stream>>>(hu1, e_src, e_dst, sum_r, E);
    sage_combine_kernel<<<gu, blk, 0, stream>>>(sum_u, deg_u, hu1, Wl2_ru, bl2_ru, Wr2_ru, out_u, U, 0);
    sage_combine_kernel<<<gr, blk, 0, stream>>>(sum_r, deg_r, hr1, Wl2_ur, bl2_ur, Wr2_ur, out_r, R, 0);
}
